// SpaceSelfAttention_8538394985248
// MI455X (gfx1250) — compile-verified, matrix operand order checked by reading, not run on hardware
//
#include <hip/hip_runtime.h>
#include <hip/hip_bf16.h>
#include <cstdint>

// ---------------------------------------------------------------------------
// SpaceSelfAttention on MI455X (gfx1250)
//   - all GEMMs via v_wmma_f32_16x16x32_bf16 (wave32 WMMA)
//   - GEMM tiles staged with the Tensor Data Mover (tensor_load_to_lds),
//     double-buffered, synchronized with s_wait_tensorcnt + barriers
//   - block tile 128x128, wave tile 32x64 (8 WMMA / K-step / wave)
// Pipeline: [cvt x][cvt+transpose weights] -> [QKV gemm] -> [attn] -> [out gemm]
// ---------------------------------------------------------------------------

typedef __attribute__((ext_vector_type(16))) __bf16 v16bf;
typedef __attribute__((ext_vector_type(8)))  __bf16 v8bf;
typedef __attribute__((ext_vector_type(8)))  float  v8f;
typedef unsigned int u32x4 __attribute__((ext_vector_type(4)));
typedef int          i32x4 __attribute__((ext_vector_type(4)));
typedef int          i32x8 __attribute__((ext_vector_type(8)));

#ifndef __has_builtin
#define __has_builtin(x) 0
#endif
#if __has_builtin(__builtin_amdgcn_tensor_load_to_lds) && \
    __has_builtin(__builtin_amdgcn_s_wait_tensorcnt)
#define HAVE_TDM 1
#else
#define HAVE_TDM 0
#endif

// Problem constants (from setup_inputs)
#define BB 4
#define TT 32
#define SS 320
#define DD 768
#define HH 12
#define HD 64
#define NL 64             // n_latents
#define D3 (3*DD)         // 2304
#define MROWS (BB*TT*SS)  // 40960

__device__ __forceinline__ v16bf frag_ld(const __bf16* base, int ld, int lane) {
    // CDNA5 16-bit A/B fragment layout (ISA 7.12.2):
    // lanes 0-15 : row/col = lane,    elems 0-7 = K 0-7,  elems 8-15 = K 16-23
    // lanes 16-31: row/col = lane-16, elems 0-7 = K 8-15, elems 8-15 = K 24-31
    int half = lane >> 4;
    int r    = lane & 15;
    const __bf16* p = base + r * ld + half * 8;
    v8bf lo = *(const v8bf*)(p);
    v8bf hi = *(const v8bf*)(p + 16);
    v16bf f;
#pragma unroll
    for (int i = 0; i < 8; ++i) { f[i] = lo[i]; f[i + 8] = hi[i]; }
    return f;
}

__device__ __forceinline__ v8f wmma_bf16(v16bf a, v16bf b, v8f c) {
    return __builtin_amdgcn_wmma_f32_16x16x32_bf16(
        /*neg_a=*/false, a, /*neg_b=*/false, b,
        /*c_mod=*/(short)0, c, /*reuse_a=*/false, /*reuse_b=*/false);
}

#if HAVE_TDM
// 2D tile DMA: global (row-major, 2-byte elems) -> LDS, per D# spec
// (CDNA5 ISA group0, §8.4 group1).
__device__ __forceinline__ void tdm_load_2d(uint32_t lds_addr, const void* gptr,
                                            uint32_t tdim0, uint32_t tdim1,
                                            uint32_t tile0, uint32_t tile1,
                                            uint64_t stride0) {
    uint64_t ga = (uint64_t)(uintptr_t)gptr;
    u32x4 g0;
    g0[0] = 1u;                                            // count=1 (valid D#)
    g0[1] = lds_addr;                                      // lds_addr[31:0]
    g0[2] = (uint32_t)ga;                                  // global_addr[31:0]
    g0[3] = (uint32_t)((ga >> 32) & 0x1FFFFFFu)            // global_addr[56:32]
          | (2u << 30);                                    // type=2 ("image")
    i32x8 g1;
    g1[0] = (int)(1u << 16);                               // data_size=1 (2B)
    g1[1] = (int)((tdim0 & 0xFFFFu) << 16);                // tensor_dim0[15:0]
    g1[2] = (int)((tdim0 >> 16) | ((tdim1 & 0xFFFFu) << 16));
    g1[3] = (int)((tdim1 >> 16) | (tile0 << 16));          // tile_dim0
    g1[4] = (int)(tile1);                                  // tile_dim1 (tile_dim2=0)
    g1[5] = (int)(uint32_t)stride0;                        // tensor_dim0_stride[31:0]
    g1[6] = (int)(uint32_t)((stride0 >> 32) & 0xFFFFu);    // stride[47:32]
    g1[7] = 0;
    i32x4 z4 = {0, 0, 0, 0};
#if __clang_major__ >= 23
    i32x8 z8 = {0, 0, 0, 0, 0, 0, 0, 0};
    __builtin_amdgcn_tensor_load_to_lds(g0, g1, z4, z4, z8, 0);
#else
    __builtin_amdgcn_tensor_load_to_lds(g0, g1, z4, z4, 0);
#endif
}
#endif

// ---------------------------------------------------------------------------
// converts
// ---------------------------------------------------------------------------
__global__ void cvt_f32_to_bf16(const float* __restrict__ in,
                                __bf16* __restrict__ out, int n) {
    int i = blockIdx.x * blockDim.x + threadIdx.x;
    if (i < n) out[i] = (__bf16)in[i];
}

// in: [K,N] f32  ->  out: [N,K] bf16  (n-major weights for TDM + B-fragments)
__global__ void cvt_transpose_f32_to_bf16(const float* __restrict__ in,
                                          __bf16* __restrict__ out,
                                          int K, int N) {
    int idx = blockIdx.x * blockDim.x + threadIdx.x;
    if (idx < K * N) {
        int k = idx / N, n = idx % N;
        out[(long)n * K + k] = (__bf16)in[idx];
    }
}

// ---------------------------------------------------------------------------
// Tiled WMMA GEMM: C[M,N] = A[M,K] * B^T[N,K] + bias[N]
// Block tile 128x128, 8 waves (4m x 2n), wave tile 32x64 (2x4 WMMA accs).
// A and BT are bf16; tiles staged by TDM, double-buffered.
// ---------------------------------------------------------------------------
template <int C_BF16>
__global__ __launch_bounds__(256)
void gemm_wmma(const __bf16* __restrict__ A, const __bf16* __restrict__ BT,
               const float* __restrict__ bias, void* __restrict__ Cp,
               int M, int N, int K) {
    __shared__ __bf16 As[2][128 * 32];   // [m][k] row-major tiles
    __shared__ __bf16 Bs[2][128 * 32];   // [n][k] n-major tiles

    const int t = threadIdx.x, lane = t & 31, wave = t >> 5;
    const int m0 = blockIdx.y * 128, n0 = blockIdx.x * 128;
    const int wm = (wave >> 1) * 32, wn = (wave & 1) * 64;
    const int nk = K / 32;

    v8f acc[2][4] = {};

#if HAVE_TDM
    if (wave == 0) {
        tdm_load_2d((uint32_t)(uintptr_t)&As[0][0], A + (long)m0 * K,
                    (uint32_t)K, (uint32_t)M, 32u, 128u, (uint64_t)K);
        tdm_load_2d((uint32_t)(uintptr_t)&Bs[0][0], BT + (long)n0 * K,
                    (uint32_t)K, (uint32_t)N, 32u, 128u, (uint64_t)K);
    }
#endif

    for (int i = 0; i < nk; ++i) {
        const int buf = i & 1;
#if HAVE_TDM
        if (wave == 0) __builtin_amdgcn_s_wait_tensorcnt(0);
        __syncthreads();                        // tiles[buf] visible to all waves
        if (wave == 0 && i + 1 < nk) {          // prefetch next K-step via TDM
            int k1 = (i + 1) * 32;
            tdm_load_2d((uint32_t)(uintptr_t)&As[buf ^ 1][0],
                        A + (long)m0 * K + k1,
                        (uint32_t)K, (uint32_t)M, 32u, 128u, (uint64_t)K);
            tdm_load_2d((uint32_t)(uintptr_t)&Bs[buf ^ 1][0],
                        BT + (long)n0 * K + k1,
                        (uint32_t)K, (uint32_t)N, 32u, 128u, (uint64_t)K);
        }
#else
        __syncthreads();
        {
            int k0 = i * 32;
#pragma unroll
            for (int j = 0; j < 16; ++j) {      // A tile: 4096 elems
                int idx = j * 256 + t;
                int r = idx >> 5, kk = idx & 31;
                As[buf][idx] = A[(long)(m0 + r) * K + k0 + kk];
            }
#pragma unroll
            for (int j = 0; j < 16; ++j) {      // B tile: 4096 elems
                int idx = j * 256 + t;
                int n = idx >> 5, kk = idx & 31;
                Bs[buf][idx] = BT[(long)(n0 + n) * K + k0 + kk];
            }
        }
        __syncthreads();
#endif
        v16bf a0 = frag_ld(&As[buf][(wm +  0) * 32], 32, lane);
        v16bf a1 = frag_ld(&As[buf][(wm + 16) * 32], 32, lane);
        v16bf b0 = frag_ld(&Bs[buf][(wn +  0) * 32], 32, lane);
        v16bf b1 = frag_ld(&Bs[buf][(wn + 16) * 32], 32, lane);
        v16bf b2 = frag_ld(&Bs[buf][(wn + 32) * 32], 32, lane);
        v16bf b3 = frag_ld(&Bs[buf][(wn + 48) * 32], 32, lane);
        acc[0][0] = wmma_bf16(a0, b0, acc[0][0]);
        acc[0][1] = wmma_bf16(a0, b1, acc[0][1]);
        acc[0][2] = wmma_bf16(a0, b2, acc[0][2]);
        acc[0][3] = wmma_bf16(a0, b3, acc[0][3]);
        acc[1][0] = wmma_bf16(a1, b0, acc[1][0]);
        acc[1][1] = wmma_bf16(a1, b1, acc[1][1]);
        acc[1][2] = wmma_bf16(a1, b2, acc[1][2]);
        acc[1][3] = wmma_bf16(a1, b3, acc[1][3]);
    }

    // epilogue: C/D layout: lanes 0-15 rows 0-7, lanes 16-31 rows 8-15
    const int col16 = lane & 15, rbase = (lane >> 4) * 8;
#pragma unroll
    for (int mi = 0; mi < 2; ++mi)
#pragma unroll
        for (int ni = 0; ni < 4; ++ni) {
            int col = n0 + wn + ni * 16 + col16;
            float bv = bias[col];
#pragma unroll
            for (int e = 0; e < 8; ++e) {
                int row = m0 + wm + mi * 16 + rbase + e;
                float v = acc[mi][ni][e] + bv;
                long o = (long)row * N + col;
                if (C_BF16) ((__bf16*)Cp)[o] = (__bf16)v;
                else        ((float*)Cp)[o]  = v;
            }
        }
}

// ---------------------------------------------------------------------------
// Block-sparse masked attention. One block = (bt, head, 32 query rows).
// LDS: sc 32x320 f32 | q 32x64 | k 320x64 | vT 64x320  (~124 KB dynamic)
// scores = q k^T / 8 (WMMA); latent rows (<64) masked vs patch cols (>=64);
// f32 softmax in LDS; attn repacked bf16 in place; y = attn * v (WMMA).
// ---------------------------------------------------------------------------
__global__ __launch_bounds__(256)
void attn_wmma(const __bf16* __restrict__ qkv, __bf16* __restrict__ y) {
    extern __shared__ char smem[];
    float*  sc = (float*)smem;                              // 32*320*4 = 40960 B
    __bf16* at = (__bf16*)smem;                             // overlay, row stride 640
    __bf16* qs = (__bf16*)(smem + 40960);                   // 32*64
    __bf16* ks = qs + 32 * 64;                              // 320*64  [kc][d]
    __bf16* vs = ks + 320 * 64;                             // 64*320  [d][kc]

    const int t = threadIdx.x, lane = t & 31, wave = t >> 5;
    const int qb = blockIdx.x, h = blockIdx.y, bt = blockIdx.z;
    const int qr0 = qb * 32;
    const long base = (long)bt * SS * D3;
    const float NEGINF = -__builtin_inff();

#pragma unroll
    for (int j = 0; j < 8; ++j) {                 // q: 2048 elems
        int idx = j * 256 + t;
        int i = idx >> 6, d = idx & 63;
        qs[idx] = qkv[base + (long)(qr0 + i) * D3 + h * HD + d];
    }
    for (int j = 0; j < 80; ++j) {                // k, v^T: 20480 elems each
        int idx = j * 256 + t;
        int kc = idx >> 6, d = idx & 63;
        ks[idx]         = qkv[base + (long)kc * D3 +     DD + h * HD + d];
        vs[d * SS + kc] = qkv[base + (long)kc * D3 + 2 * DD + h * HD + d];
    }
    __syncthreads();

    // scores: 2 qtiles x 20 ktiles = 40 tiles over 8 waves
    for (int tt = wave; tt < 40; tt += 8) {
        int qt = tt / 20, kt = tt % 20;
        v8f a = {};
        a = wmma_bf16(frag_ld(qs + qt * 16 * 64,      64, lane),
                      frag_ld(ks + kt * 16 * 64,      64, lane), a);
        a = wmma_bf16(frag_ld(qs + qt * 16 * 64 + 32, 64, lane),
                      frag_ld(ks + kt * 16 * 64 + 32, 64, lane), a);
        int col = kt * 16 + (lane & 15);
        int rb = (lane >> 4) * 8;
        bool maskcol = (qr0 < NL) && (col >= NL);   // latent query, patch key
#pragma unroll
        for (int e = 0; e < 8; ++e) {
            int r = qt * 16 + rb + e;
            sc[r * SS + col] = maskcol ? NEGINF : a[e] * 0.125f;
        }
    }
    __syncthreads();

    // softmax: one row per lane (threads 0-31), f32 in LDS
    if (t < 32) {
        int r = t;
        float m = NEGINF;
        for (int kc = 0; kc < SS; ++kc) m = fmaxf(m, sc[r * SS + kc]);
        float s = 0.f;
        for (int kc = 0; kc < SS; ++kc) {
            float e = __expf(sc[r * SS + kc] - m);
            s += e;
            sc[r * SS + kc] = e;
        }
        float inv = 1.f / s;
        // in-place bf16 repack: write offset 2*kc <= read offset 4*kc (safe)
        for (int kc = 0; kc < SS; ++kc)
            at[r * 640 + kc] = (__bf16)(sc[r * SS + kc] * inv);
    }
    __syncthreads();

    // y = attn * v : 2 qtiles x 4 dtiles = 8 tiles, one per wave
    {
        int qt = wave >> 2, dt = wave & 3;
        v8f acc = {};
        for (int kc0 = 0; kc0 < SS; kc0 += 32)
            acc = wmma_bf16(frag_ld(at + qt * 16 * 640 + kc0, 640, lane),
                            frag_ld(vs + dt * 16 * SS  + kc0, SS,  lane), acc);
        int col = dt * 16 + (lane & 15);
        int rb = (lane >> 4) * 8;
#pragma unroll
        for (int e = 0; e < 8; ++e) {
            int r = qr0 + qt * 16 + rb + e;
            y[((long)bt * SS + r) * DD + h * HD + col] = (__bf16)acc[e];
        }
    }
}

// ---------------------------------------------------------------------------
extern "C" void kernel_launch(void* const* d_in, const int* in_sizes, int n_in,
                              void* d_out, int out_size, void* d_ws, size_t ws_size,
                              hipStream_t stream) {
    const float* x      = (const float*)d_in[0];   // [B,T,S,D]
    const float* qkv_w  = (const float*)d_in[1];   // [D,3D]
    const float* qkv_b  = (const float*)d_in[2];   // [3D]
    const float* out_w  = (const float*)d_in[3];   // [D,D]
    const float* out_b  = (const float*)d_in[4];   // [D]
    float* out = (float*)d_out;

    // workspace carve-up (bf16). xb is dead after the QKV GEMM, so the
    // attention output y aliases it (same size MROWS*DD).
    char* w = (char*)d_ws;
    __bf16* xb      = (__bf16*)w;                 w += (size_t)MROWS * DD * 2;
    __bf16* yb      = xb;                         // alias (see above)
    __bf16* qkvwT   = (__bf16*)w;                 w += (size_t)D3 * DD * 2;
    __bf16* outwT   = (__bf16*)w;                 w += (size_t)DD * DD * 2;
    __bf16* qkv     = (__bf16*)w;                 w += (size_t)MROWS * D3 * 2;

    // 1) conversions: x -> bf16, weights -> transposed bf16 [N][K]
    {
        int nx = MROWS * DD;
        cvt_f32_to_bf16<<<(nx + 255) / 256, 256, 0, stream>>>(x, xb, nx);
        int n1 = DD * D3;
        cvt_transpose_f32_to_bf16<<<(n1 + 255) / 256, 256, 0, stream>>>(
            qkv_w, qkvwT, DD, D3);
        int n2 = DD * DD;
        cvt_transpose_f32_to_bf16<<<(n2 + 255) / 256, 256, 0, stream>>>(
            out_w, outwT, DD, DD);
    }
    // 2) QKV projection: [40960,768] x [768,2304] -> bf16 qkv
    {
        dim3 grid(D3 / 128, MROWS / 128);
        gemm_wmma<1><<<grid, 256, 0, stream>>>(xb, qkvwT, qkv_b, qkv,
                                               MROWS, D3, DD);
    }
    // 3) masked attention: grid = (query blocks, heads, B*T)
    {
        dim3 grid(SS / 32, HH, BB * TT);
        size_t smem = 40960 + (32 * 64 + 320 * 64 + 64 * 320) * sizeof(__bf16);
        attn_wmma<<<grid, 256, smem, stream>>>(qkv, yb);
    }
    // 4) output projection: [40960,768] x [768,768] -> f32 out
    {
        dim3 grid(DD / 128, MROWS / 128);
        gemm_wmma<0><<<grid, 256, 0, stream>>>(yb, outwT, out_b, out,
                                               MROWS, DD, DD);
    }
}
